// Agent_13116830122193
// MI455X (gfx1250) — matrix operand order repaired and checked by reading, not compiled, not run on hardware
//
#include <hip/hip_runtime.h>
#include <hip/hip_bf16.h>

// ---------------- constants ----------------
#define TT   16384          // time steps
#define HH   256            // hidden
#define FH   1024           // 4*H (gates)
#define FF   144            // conv features
#define FP   160            // padded K for WMMA (5 tiles of 32)
#define AA   15             // actions

typedef __attribute__((ext_vector_type(16))) __bf16 bf16x16;
typedef __attribute__((ext_vector_type(8)))  float  f32x8;

#if defined(__has_builtin)
#if __has_builtin(__builtin_amdgcn_tensor_load_to_lds) && __has_builtin(__builtin_amdgcn_s_wait_tensorcnt)
#define USE_TDM 1
#endif
#endif

#ifdef USE_TDM
typedef __attribute__((ext_vector_type(4))) unsigned int u32x4;
typedef __attribute__((ext_vector_type(8))) int          i32x8;
typedef __attribute__((ext_vector_type(4))) int          i32x4;

// TDM: move `nelem` consecutive fp32 (1D tile) from global to LDS.
static __device__ __forceinline__ void tdm_load_row_f32(unsigned lds_byte_addr,
                                                        const float* gaddr,
                                                        unsigned nelem) {
    unsigned long long ga = (unsigned long long)(const void*)gaddr;
    u32x4 g0;
    g0[0] = 1u;                                   // count=1, user mode
    g0[1] = lds_byte_addr;                        // lds_addr (bytes)
    g0[2] = (unsigned)ga;                         // global_addr[31:0]
    g0[3] = ((unsigned)(ga >> 32) & 0x01FFFFFFu)  // global_addr[56:32]
          | 0x80000000u;                          // type=2 ("image") at bits 127:126
    i32x8 g1;
    g1[0] = (int)(2u << 16);                      // data_size=2 (4B); wg_mask=0
    g1[1] = (int)((nelem & 0xFFFFu) << 16);       // tensor_dim0[15:0] -> bits 63:48
    g1[2] = (int)((nelem >> 16) | (1u << 16));    // tensor_dim0[31:16], tensor_dim1=1
    g1[3] = (int)((nelem & 0xFFFFu) << 16);       // tile_dim0 -> bits 127:112
    g1[4] = 0;                                    // tile_dim1=0 (unused), tile_dim2=0
    g1[5] = (int)nelem;                           // tensor_dim0_stride[31:0]
    g1[6] = 0;
    g1[7] = 0;
    i32x4 gz = {0, 0, 0, 0};                      // groups 2/3 unused (<=2D)
#if __clang_major__ >= 23
    i32x8 gz8 = {0, 0, 0, 0, 0, 0, 0, 0};
    __builtin_amdgcn_tensor_load_to_lds(g0, g1, gz, gz, gz8, 0);
#else
    __builtin_amdgcn_tensor_load_to_lds(g0, g1, gz, gz, 0);
#endif
}
#endif // USE_TDM

static __device__ __forceinline__ unsigned short f2bf(float f) {
    unsigned int u = __float_as_uint(f);
    unsigned int r = u + 0x7FFFu + ((u >> 16) & 1u);   // RNE
    return (unsigned short)(r >> 16);
}

static __device__ __forceinline__ bf16x16 mk_frag(uint4 lo, uint4 hi) {
    union { struct { uint4 a, b; } u; bf16x16 v; } x;
    x.u.a = lo; x.u.b = hi;
    return x.v;
}

static __device__ __forceinline__ f32x8 wmma_bf16(bf16x16 a, bf16x16 b, f32x8 c) {
    // D = A(16x32 bf16) * B(32x16 bf16) + C(16x16 f32)
    return __builtin_amdgcn_wmma_f32_16x16x32_bf16(false, a, false, b, (short)0, c, false, false);
}

static __device__ __forceinline__ float sigmoidf_(float x) {
    return 1.0f / (1.0f + __expf(-x));
}

// ---------------- kernel 0: weight prep (fp32 -> bf16, padded) ----------------
__global__ void prep_kernel(const float* __restrict__ w_ih,  // [1024,144]
                            const float* __restrict__ w_hh,  // [1024,256]
                            const float* __restrict__ b_ih,
                            const float* __restrict__ b_hh,
                            unsigned short* __restrict__ w_ihB,  // [1024,160] bf16
                            unsigned short* __restrict__ w_hhB,  // [1024,256] bf16
                            float* __restrict__ bias)            // [1024]
{
    int idx = blockIdx.x * blockDim.x + threadIdx.x;
    if (idx < FH * FP) {
        int row = idx / FP, k = idx % FP;
        w_ihB[idx] = (k < FF) ? f2bf(w_ih[row * FF + k]) : (unsigned short)0;
    }
    if (idx < FH * HH) {
        w_hhB[idx] = f2bf(w_hh[idx]);
    }
    if (idx < FH) {
        bias[idx] = b_ih[idx] + b_hh[idx];
    }
}

// ---------------- kernel 1: conv encoder + xg GEMM via WMMA ----------------
// block = 256 threads (8 waves); each block handles 16 time rows x all 1024 cols.
__global__ __launch_bounds__(256) void encode_xg_kernel(
        const float* __restrict__ x,        // [T,4,5,5]
        const float* __restrict__ conv_w,   // [16,4,3,3]
        const float* __restrict__ conv_b,   // [16]
        const unsigned short* __restrict__ w_ihB,  // [1024,160] bf16 (K contiguous)
        const float* __restrict__ bias,     // [1024]
        float* __restrict__ xg)             // [T,1024]
{
    __shared__ float xs[16][100];                            // 6.4 KB
    __shared__ __align__(16) unsigned short featB[16][FP];   // 5 KB, bf16 padded

    const int tid = threadIdx.x;
    const int t0  = blockIdx.x * 16;

    // stage x / 255
    for (int i = tid; i < 16 * 100; i += 256) {
        int tl = i / 100, r = i % 100;
        xs[tl][r] = x[(t0 + tl) * 100 + r] * (1.0f / 255.0f);
    }
    // zero K padding 144..159
    for (int i = tid; i < 16 * (FP - FF); i += 256) {
        featB[i / (FP - FF)][FF + (i % (FP - FF))] = 0;
    }
    __syncthreads();

    // conv 3x3 VALID + relu: thread -> (t-local, out-channel), 3x3 spatial each
    {
        int tl = tid >> 4, oc = tid & 15;
        float b = conv_b[oc];
        #pragma unroll
        for (int i = 0; i < 3; ++i)
            #pragma unroll
            for (int j = 0; j < 3; ++j) {
                float acc = b;
                #pragma unroll
                for (int c = 0; c < 4; ++c)
                    #pragma unroll
                    for (int ki = 0; ki < 3; ++ki)
                        #pragma unroll
                        for (int kj = 0; kj < 3; ++kj)
                            acc += xs[tl][c * 25 + (i + ki) * 5 + (j + kj)] *
                                   conv_w[(oc * 4 + c) * 9 + ki * 3 + kj];
                acc = fmaxf(acc, 0.0f);
                featB[tl][oc * 9 + i * 3 + j] = f2bf(acc);
            }
    }
    __syncthreads();

    // GEMM: xg[16 rows, 1024 cols] = featB(16x160) @ w_ihB^T + bias
    const int wave = tid >> 5, lane = tid & 31;
    const int half = lane >> 4, ln = lane & 15;

    // A fragments from LDS (row = ln; half picks contiguous K half, matching B)
    bf16x16 afrag[5];
    #pragma unroll
    for (int kt = 0; kt < 5; ++kt) {
        int kb = kt * 32 + half * 16;
        uint4 lo = *(const uint4*)&featB[ln][kb];
        uint4 hi = *(const uint4*)&featB[ln][kb + 8];
        afrag[kt] = mk_frag(lo, hi);
    }

    const int colBase = wave * 128;    // 8 waves x 128 columns
    #pragma unroll
    for (int nt = 0; nt < 8; ++nt) {
        int col = colBase + nt * 16 + ln;
        f32x8 acc = {};
        #pragma unroll
        for (int kt = 0; kt < 5; ++kt) {
            int kb = kt * 32 + half * 16;          // lanes 0-15: K lo half; 16-31: K hi half
            const uint4* p = (const uint4*)&w_ihB[col * FP + kb];
            bf16x16 bfrag = mk_frag(p[0], p[1]);
            acc = wmma_bf16(afrag[kt], bfrag, acc);
        }
        float bv = bias[col];
        #pragma unroll
        for (int v = 0; v < 8; ++v) {
            int row = v + half * 8;                // C layout: vgpr v, half selects M or M+8
            xg[(size_t)(t0 + row) * FH + col] = acc[v] + bv;
        }
    }
}

// ---------------- kernel 2: persistent single-WGP LSTM scan ----------------
// 1024 threads = 32 wave32s on one WGP. w_hh^T resident in VGPRs (bf16),
// h broadcast through 512B of LDS, gates gathered through 4KB of LDS.
// xg rows (4KB/step) staged one step ahead into LDS by the Tensor Data Mover.
// 2 barriers per step: done-reset is folded into the previous step's epilogue.
__global__ __launch_bounds__(1024) void lstm_scan_kernel(
        const float* __restrict__ xg,        // [T,1024]
        const unsigned short* __restrict__ w_hhB, // [1024,256] bf16
        const int* __restrict__ done,        // [T]
        const float* __restrict__ h0,
        const float* __restrict__ c0,
        float* __restrict__ hs,              // [T,256]
        float* __restrict__ out_tail)        // hN(256), cN(256)
{
    __shared__ __align__(16) unsigned short hB[HH];  // h in bf16 (reset-applied)
    __shared__ float gates[FH];
#ifdef USE_TDM
    __shared__ __align__(16) float xls[2][FH];       // double-buffered xg row
#endif

    const int tid  = threadIdx.x;
    const int wave = tid >> 5, lane = tid & 31;
    const int half = lane >> 4, ln = lane & 15;
    const int colBase = wave * 32;                   // each wave owns 32 gate columns

    // Preload resident B fragments: 2 N-tiles x 8 K-tiles x 32B/lane = 128 VGPRs/lane.
    uint4 bfr[2][8][2];
    #pragma unroll
    for (int c = 0; c < 2; ++c) {
        int col = colBase + c * 16 + ln;
        #pragma unroll
        for (int kt = 0; kt < 8; ++kt) {
            int kb = kt * 32 + half * 16;
            const uint4* p = (const uint4*)&w_hhB[col * HH + kb];
            bfr[c][kt][0] = p[0];
            bfr[c][kt][1] = p[1];
        }
    }

    float cstate = 0.0f;
    {
        float nd0 = done[0] ? 0.0f : 1.0f;           // reset BEFORE first matvec
        if (tid < HH) {
            cstate = c0[tid] * nd0;
            hB[tid] = f2bf(h0[tid] * nd0);
        }
    }
#ifdef USE_TDM
    if (wave == 0) {                                 // stage xg row for t=0
        tdm_load_row_f32((unsigned)(uintptr_t)&xls[0][0], xg, FH);
        __builtin_amdgcn_s_wait_tensorcnt(0);
    }
#endif
    __syncthreads();

    const int xcol0 = colBase + ln;
    const int xcol1 = colBase + 16 + ln;

    for (int t = 0; t < TT; ++t) {
#ifdef USE_TDM
        // async-stage next step's xg row while this step's WMMAs run
        if (wave == 0 && t + 1 < TT) {
            tdm_load_row_f32((unsigned)(uintptr_t)&xls[(t + 1) & 1][0],
                             xg + (size_t)(t + 1) * FH, FH);
        }
#else
        if (half == 0 && t + 1 < TT) {
            __builtin_prefetch(&xg[(size_t)(t + 1) * FH + xcol0], 0, 3);
        }
#endif

        // gates = h @ w_hh^T  (A = h broadcast into all 16 rows)
        f32x8 acc0 = {}, acc1 = {};
        #pragma unroll
        for (int kt = 0; kt < 8; ++kt) {
            int kb = kt * 32 + half * 16;
            uint4 lo = *(const uint4*)&hB[kb];
            uint4 hi = *(const uint4*)&hB[kb + 8];
            bf16x16 af = mk_frag(lo, hi);
            acc0 = wmma_bf16(af, mk_frag(bfr[0][kt][0], bfr[0][kt][1]), acc0);
            acc1 = wmma_bf16(af, mk_frag(bfr[1][kt][0], bfr[1][kt][1]), acc1);
        }
        // every row of D is identical; lanes 0-15 publish their column's gate
        if (half == 0) {
#ifdef USE_TDM
            gates[xcol0] = acc0[0] + xls[t & 1][xcol0];
            gates[xcol1] = acc1[0] + xls[t & 1][xcol1];
#else
            gates[xcol0] = acc0[0] + xg[(size_t)t * FH + xcol0];
            gates[xcol1] = acc1[0] + xg[(size_t)t * FH + xcol1];
#endif
        }
        __syncthreads();   // gates visible

        // LSTM cell (torch gate order i,f,g,o); fold next step's done-reset in
        if (tid < HH) {
            float gi = gates[tid];
            float gf = gates[HH + tid];
            float gg = gates[2 * HH + tid];
            float go = gates[3 * HH + tid];
            float cc = sigmoidf_(gf) * cstate + sigmoidf_(gi) * tanhf(gg);
            float h  = sigmoidf_(go) * tanhf(cc);
            hs[(size_t)t * HH + tid] = h;            // unscaled post-step h
            if (t == TT - 1) {
                out_tail[tid]      = h;              // hN
                out_tail[HH + tid] = cc;             // cN
            }
            float ndn = 1.0f;
            if (t + 1 < TT) ndn = done[t + 1] ? 0.0f : 1.0f;
            cstate = cc * ndn;
            hB[tid] = f2bf(h * ndn);                 // reset-applied h for next matvec
        }
#ifdef USE_TDM
        if (wave == 0) {                             // next xg buffer filled before the
            __builtin_amdgcn_s_wait_tensorcnt(0);    // barrier that publishes it
        }
#endif
        __syncthreads();   // hB (+ staged xg) ready for step t+1
    }
}

// ---------------- kernel 3: MLP + actor/critic heads + log-softmax ----------------
__global__ __launch_bounds__(256) void heads_kernel(
        const float* __restrict__ hs,        // [T,256]
        const float* __restrict__ mlp_w,     // [32,256]
        const float* __restrict__ mlp_b,     // [32]
        const float* __restrict__ actor_w,   // [15,32]
        const float* __restrict__ actor_b,   // [15]
        const float* __restrict__ critic_w,  // [1,32]
        const float* __restrict__ critic_b,  // [1]
        const int* __restrict__ action,      // [T]
        float* __restrict__ out)             // logprob[T], entropy[T], value[T]
{
    __shared__ float Wm[32 * HH];
    __shared__ float Wa[AA * 32];
    __shared__ float Bm[32], Wc[32], Ba[AA], Bc;

    const int tid = threadIdx.x;
    for (int i = tid; i < 32 * HH; i += 256) Wm[i] = mlp_w[i];
    for (int i = tid; i < AA * 32; i += 256) Wa[i] = actor_w[i];
    if (tid < 32) { Bm[tid] = mlp_b[tid]; Wc[tid] = critic_w[tid]; }
    if (tid < AA) Ba[tid] = actor_b[tid];
    if (tid == 0) Bc = critic_b[0];
    __syncthreads();

    const int t = blockIdx.x * 256 + tid;

    float z[32];
    #pragma unroll
    for (int j = 0; j < 32; ++j) z[j] = Bm[j];
    for (int k = 0; k < HH; ++k) {
        float hv = fmaxf(hs[(size_t)t * HH + k], 0.0f);  // relu(hs)
        #pragma unroll
        for (int j = 0; j < 32; ++j) z[j] += hv * Wm[j * HH + k];
    }
    #pragma unroll
    for (int j = 0; j < 32; ++j) z[j] = fmaxf(z[j], 0.0f);

    float logits[AA];
    float m = -1e30f;
    #pragma unroll
    for (int a = 0; a < AA; ++a) {
        float s = Ba[a];
        #pragma unroll
        for (int j = 0; j < 32; ++j) s += z[j] * Wa[a * 32 + j];
        logits[a] = s;
        m = fmaxf(m, s);
    }
    float se = 0.0f;
    #pragma unroll
    for (int a = 0; a < AA; ++a) se += __expf(logits[a] - m);
    float lse = __logf(se) + m;

    float ent = 0.0f;
    #pragma unroll
    for (int a = 0; a < AA; ++a) {
        float lp = logits[a] - lse;
        ent -= __expf(lp) * lp;
    }
    int act = action[t];
    float logprob = logits[act] - lse;

    float v = Bc;
    #pragma unroll
    for (int j = 0; j < 32; ++j) v += z[j] * Wc[j];

    out[t]           = logprob;
    out[TT + t]      = ent;
    out[2 * TT + t]  = v;
}

// ---------------- launcher ----------------
extern "C" void kernel_launch(void* const* d_in, const int* in_sizes, int n_in,
                              void* d_out, int out_size, void* d_ws, size_t ws_size,
                              hipStream_t stream) {
    // setup_inputs order
    const float* x        = (const float*)d_in[0];
    const float* h0       = (const float*)d_in[1];
    const float* c0       = (const float*)d_in[2];
    const float* conv_w   = (const float*)d_in[3];
    const float* conv_b   = (const float*)d_in[4];
    const float* w_ih     = (const float*)d_in[5];
    const float* w_hh     = (const float*)d_in[6];
    const float* b_ih     = (const float*)d_in[7];
    const float* b_hh     = (const float*)d_in[8];
    const float* mlp_w    = (const float*)d_in[9];
    const float* mlp_b    = (const float*)d_in[10];
    const float* actor_w  = (const float*)d_in[11];
    const float* actor_b  = (const float*)d_in[12];
    const float* critic_w = (const float*)d_in[13];
    const float* critic_b = (const float*)d_in[14];
    const int*   done     = (const int*)d_in[15];
    const int*   action   = (const int*)d_in[16];

    float* out = (float*)d_out;

    // workspace layout
    char* ws = (char*)d_ws;
    unsigned short* w_ihB = (unsigned short*)(ws);                      // 1024*160*2 = 327680
    unsigned short* w_hhB = (unsigned short*)(ws + 327680);             // 1024*256*2 = 524288
    float*          bias  = (float*)(ws + 327680 + 524288);             // 4096
    float*          xg    = (float*)(ws + 856064);                      // 16384*1024*4 = 64 MB
    float*          hs    = (float*)(ws + 856064 + (size_t)TT * FH * 4);// 16384*256*4  = 16 MB

    // 0) weight prep
    prep_kernel<<<(FH * HH + 255) / 256, 256, 0, stream>>>(
        w_ih, w_hh, b_ih, b_hh, w_ihB, w_hhB, bias);

    // 1) conv encoder + xg GEMM (WMMA bf16)
    encode_xg_kernel<<<TT / 16, 256, 0, stream>>>(
        x, conv_w, conv_b, w_ihB, bias, xg);

    // 2) sequential LSTM scan: one persistent workgroup, w_hh resident in VGPRs
    lstm_scan_kernel<<<1, 1024, 0, stream>>>(
        xg, w_hhB, done, h0, c0, hs, out + 3 * TT);

    // 3) heads
    heads_kernel<<<TT / 256, 256, 0, stream>>>(
        hs, mlp_w, mlp_b, actor_w, actor_b, critic_w, critic_b, action, out);
}
